// MultichannelMultiheadAttention_72971494359081
// MI455X (gfx1250) — compile-verified, matrix operand order checked by reading, not run on hardware
//
#include <hip/hip_runtime.h>

#define BINS  1024
#define SEQ   1024
#define NHEAD 8
#define HDIM  128
#define ROTD  64
#define NBC   4

typedef __attribute__((ext_vector_type(16))) __bf16 bf16x16;
typedef __attribute__((ext_vector_type(8)))  __bf16 bf16x8;
typedef __attribute__((ext_vector_type(8)))  float  f32x8;

__device__ __forceinline__ __bf16 to_bf16(float f) {
  unsigned u = __builtin_bit_cast(unsigned, f);
  u += 0x7FFFu + ((u >> 16) & 1u);            // round-to-nearest-even
  unsigned short s = (unsigned short)(u >> 16);
  return __builtin_bit_cast(__bf16, s);
}

__device__ __forceinline__ f32x8 fzero8() {
  f32x8 z = {0.f,0.f,0.f,0.f,0.f,0.f,0.f,0.f};
  return z;
}

__device__ __forceinline__ f32x8 wmma_bf16(bf16x16 a, bf16x16 b, f32x8 c) {
  // D = A(16x32) * B(32x16) + C, fp32 accumulate
  return __builtin_amdgcn_wmma_f32_16x16x32_bf16(false, a, false, b, (short)0, c, false, false);
}

// Load one 16x32 bf16 fragment (A-operand layout; B-operand of X*W^T uses the
// identical per-lane pattern since lane n of B holds W-row n along K).
// ISA layout: lane (r = lane&15, h = lane>>4) holds row r, K = [h*8, h*8+8) and
// [16+h*8, 16+h*8+8) -> two contiguous 16-byte loads.
__device__ __forceinline__ bf16x16 load_frag(const __bf16* base, int ld, int lane) {
  int r = lane & 15;
  int h = (lane >> 4) & 1;
  const __bf16* p = base + (size_t)r * ld + h * 8;
  bf16x8 lo = *(const bf16x8*)(p);
  bf16x8 hi = *(const bf16x8*)(p + 16);
  return __builtin_shufflevector(lo, hi, 0,1,2,3,4,5,6,7,8,9,10,11,12,13,14,15);
}

// ---------------------------------------------------------------- convert
__global__ void f32_to_bf16_kernel(const float* __restrict__ in,
                                   __bf16* __restrict__ out, int n) {
  int i = blockIdx.x * blockDim.x + threadIdx.x;
  if (i < n) out[i] = to_bf16(in[i]);
}

// ---------------------------------------------------------------- GEMM: Y = A * B^T
// A: [z][M][K] bf16 row-major, B: [N][K] bf16 row-major (shared over z), Y: [z][M][N] f32
// Block: 256 threads = 8 waves, 2(M) x 4(N); wave tile 32x64 -> block tile 64 x 256.
__global__ __launch_bounds__(256)
void gemm_xwT_kernel(const __bf16* __restrict__ A, const __bf16* __restrict__ B,
                     float* __restrict__ Y, long aStride, long yStride, int K, int N) {
  const int lane = threadIdx.x & 31;
  const int wave = threadIdx.x >> 5;
  const int wm = wave & 1, wn = wave >> 1;
  const int mBase = blockIdx.y * 64  + wm * 32;
  const int nBase = blockIdx.x * 256 + wn * 64;
  const __bf16* Am = A + (size_t)blockIdx.z * aStride + (size_t)mBase * K;
  const __bf16* Bn = B + (size_t)nBase * K;

  f32x8 acc[2][4];
  #pragma unroll
  for (int ms = 0; ms < 2; ++ms)
    #pragma unroll
    for (int ns = 0; ns < 4; ++ns) acc[ms][ns] = fzero8();

  for (int k0 = 0; k0 < K; k0 += 32) {
    __builtin_prefetch(Am + k0 + 512, 0, 0);   // global_prefetch_b8
    __builtin_prefetch(Bn + k0 + 512, 0, 0);
    bf16x16 a0 = load_frag(Am + k0,                  K, lane);
    bf16x16 a1 = load_frag(Am + (size_t)16 * K + k0, K, lane);
    bf16x16 b[4];
    #pragma unroll
    for (int ns = 0; ns < 4; ++ns)
      b[ns] = load_frag(Bn + (size_t)(ns * 16) * K + k0, K, lane);
    #pragma unroll
    for (int ns = 0; ns < 4; ++ns) {
      acc[0][ns] = wmma_bf16(a0, b[ns], acc[0][ns]);
      acc[1][ns] = wmma_bf16(a1, b[ns], acc[1][ns]);
    }
  }

  const int half = lane >> 4, col = lane & 15;
  float* Yp = Y + (size_t)blockIdx.z * yStride;
  #pragma unroll
  for (int ms = 0; ms < 2; ++ms)
    #pragma unroll
    for (int ns = 0; ns < 4; ++ns)
      #pragma unroll
      for (int j = 0; j < 8; ++j)
        Yp[(size_t)(mBase + ms * 16 + j + 8 * half) * N + (nBase + ns * 16 + col)] =
            acc[ms][ns][j];
}

// ---------------------------------------------------------------- conv3 + RoPE + relayout
// y: [bc][w][g] f32.  out (bf16): transposeOut=0 -> [bc*8+head][w][d] ; =1 -> [bc*8+head][d][w]
__global__ void conv_rope_kernel(const float* __restrict__ y, const float* __restrict__ ck,
                                 const float* __restrict__ invf, __bf16* __restrict__ out,
                                 int do_rope, int transposeOut) {
  int idx = blockIdx.x * blockDim.x + threadIdx.x;
  int g  = idx & (BINS - 1);
  int w  = (idx >> 10) & (SEQ - 1);
  int bc = idx >> 20;
  auto cv = [&](int gg) -> float {               // cross-correlation, zero-padded
    size_t base = (size_t)bc * SEQ * BINS + gg;
    float r = y[base + (size_t)w * BINS] * ck[BINS + gg];
    if (w > 0)       r += y[base + (size_t)(w - 1) * BINS] * ck[gg];
    if (w < SEQ - 1) r += y[base + (size_t)(w + 1) * BINS] * ck[2 * BINS + gg];
    return r;
  };
  float v0 = cv(g);
  int head = g >> 7, d = g & (HDIM - 1);
  float val = v0;
  if (do_rope && d < ROTD) {
    float vp = cv(g ^ 1);                        // rotation partner
    float ang = (float)w * invf[d >> 1];
    float s, c;
    __sincosf(ang, &s, &c);
    val = (d & 1) ? (v0 * c + vp * s) : (v0 * c - vp * s);
  }
  size_t fh = (size_t)bc * NHEAD + head;
  size_t o  = transposeOut ? ((fh * HDIM + d) * SEQ + w)
                           : ((fh * SEQ + w) * HDIM + d);
  out[o] = to_bf16(val);
}

// ---------------------------------------------------------------- flash attention
// One wave handles a 16-query tile; online softmax over 32-key chunks.
__global__ __launch_bounds__(32)
void attn_fa_kernel(const __bf16* __restrict__ qb, const __bf16* __restrict__ kb,
                    const __bf16* __restrict__ vtb, __bf16* __restrict__ ob) {
  __shared__ __align__(16) __bf16 pT[16 * 32];   // P staging: C-layout -> A-layout
  const int lane = threadIdx.x;
  const int half = lane >> 4, col = lane & 15;
  const int f  = blockIdx.y;                      // bc*8 + head
  const int qt = blockIdx.x;                      // query tile (16 rows)
  const __bf16* q  = qb  + (size_t)f * SEQ * HDIM + (size_t)qt * 16 * HDIM;
  const __bf16* k  = kb  + (size_t)f * SEQ * HDIM;
  const __bf16* vt = vtb + (size_t)f * HDIM * SEQ;

  bf16x16 qf[4];
  #pragma unroll
  for (int c = 0; c < 4; ++c) qf[c] = load_frag(q + c * 32, HDIM, lane);

  float m_i[8], l_i[8];
  f32x8 acc[8];
  #pragma unroll
  for (int j = 0; j < 8; ++j) { m_i[j] = -3.0e38f; l_i[j] = 0.f; }
  #pragma unroll
  for (int o = 0; o < 8; ++o) acc[o] = fzero8();

  for (int kt = 0; kt < SEQ; kt += 32) {
    f32x8 s0 = fzero8(), s1 = fzero8();
    #pragma unroll
    for (int c = 0; c < 4; ++c) {                 // S = q (16x128) . k^T (two 16-key tiles)
      bf16x16 kf0 = load_frag(k + (size_t)kt * HDIM + c * 32,        HDIM, lane);
      bf16x16 kf1 = load_frag(k + (size_t)(kt + 16) * HDIM + c * 32, HDIM, lane);
      s0 = wmma_bf16(qf[c], kf0, s0);
      s1 = wmma_bf16(qf[c], kf1, s1);
    }
    const float scale = 0.03125f;                 // 1/sqrt(BINS)
    float sv0[8], sv1[8], mx[8];
    #pragma unroll
    for (int j = 0; j < 8; ++j) {
      sv0[j] = s0[j] * scale;
      sv1[j] = s1[j] * scale;
      mx[j]  = fmaxf(sv0[j], sv1[j]);
    }
    #pragma unroll
    for (int off = 1; off < 16; off <<= 1) {      // row max within 16-lane half
      #pragma unroll
      for (int j = 0; j < 8; ++j) mx[j] = fmaxf(mx[j], __shfl_xor(mx[j], off, 32));
    }
    float p0[8], p1[8], alpha[8], rs[8];
    #pragma unroll
    for (int j = 0; j < 8; ++j) {
      float mnew = fmaxf(m_i[j], mx[j]);
      alpha[j] = __expf(m_i[j] - mnew);
      m_i[j]   = mnew;
      p0[j] = __expf(sv0[j] - mnew);
      p1[j] = __expf(sv1[j] - mnew);
      rs[j] = p0[j] + p1[j];
    }
    #pragma unroll
    for (int off = 1; off < 16; off <<= 1) {      // row sum within 16-lane half
      #pragma unroll
      for (int j = 0; j < 8; ++j) rs[j] += __shfl_xor(rs[j], off, 32);
    }
    #pragma unroll
    for (int j = 0; j < 8; ++j) l_i[j] = l_i[j] * alpha[j] + rs[j];
    #pragma unroll
    for (int o = 0; o < 8; ++o)
      #pragma unroll
      for (int j = 0; j < 8; ++j) acc[o][j] *= alpha[j];

    __syncthreads();                              // guard previous pf reads
    #pragma unroll
    for (int j = 0; j < 8; ++j) {                 // stage P (C-layout) to LDS
      pT[(j + 8 * half) * 32 + col]      = to_bf16(p0[j]);
      pT[(j + 8 * half) * 32 + 16 + col] = to_bf16(p1[j]);
    }
    __syncthreads();
    bf16x16 pf = load_frag((const __bf16*)pT, 32, lane);   // read back in A-layout
    #pragma unroll
    for (int o = 0; o < 8; ++o) {                 // O += P (16x32) . V (32x128)
      bf16x16 vf = load_frag(vt + (size_t)(o * 16) * SEQ + kt, SEQ, lane);
      acc[o] = wmma_bf16(pf, vf, acc[o]);
    }
  }

  const int bc = f >> 3, head = f & 7;
  #pragma unroll
  for (int j = 0; j < 8; ++j) {                   // normalize + relayout to [bc][w][g]
    float inv = 1.0f / l_i[j];
    int w = qt * 16 + j + 8 * half;
    size_t rb = ((size_t)bc * SEQ + w) * BINS + (size_t)head * HDIM;
    #pragma unroll
    for (int o = 0; o < 8; ++o)
      ob[rb + o * 16 + col] = to_bf16(acc[o][j] * inv);
  }
}

// ---------------------------------------------------------------- launch
extern "C" void kernel_launch(void* const* d_in, const int* in_sizes, int n_in,
                              void* d_out, int out_size, void* d_ws, size_t ws_size,
                              hipStream_t stream) {
  const float* x    = (const float*)d_in[0];
  const float* q_w  = (const float*)d_in[1];
  const float* q_c  = (const float*)d_in[2];
  const float* k_w  = (const float*)d_in[3];
  const float* k_c  = (const float*)d_in[4];
  const float* v_w  = (const float*)d_in[5];
  const float* v_c  = (const float*)d_in[6];
  const float* o_w  = (const float*)d_in[7];
  const float* invf = (const float*)d_in[8];

  char* ws = (char*)d_ws;
  const size_t MB = 1ull << 20;
  __bf16* xb  = (__bf16*)(ws + 0);         // 8 MB  x in bf16
  __bf16* wqb = (__bf16*)(ws + 8  * MB);   // 2 MB each weight
  __bf16* wkb = (__bf16*)(ws + 10 * MB);
  __bf16* wvb = (__bf16*)(ws + 12 * MB);
  __bf16* wob = (__bf16*)(ws + 14 * MB);
  float*  ysc = (float*)(ws + 16 * MB);    // 16 MB f32 projection scratch (reused q/k/v)
  __bf16* qbf = (__bf16*)(ws + 32 * MB);   // 8 MB [f][w][d]
  __bf16* kbf = (__bf16*)(ws + 40 * MB);   // 8 MB [f][w][d]
  __bf16* vtb = (__bf16*)(ws + 48 * MB);   // 8 MB [f][d][w] (transposed)
  __bf16* obf = (__bf16*)(ws + 56 * MB);   // 8 MB attention out, bf16  (total 64 MB)

  const int NX = NBC * SEQ * BINS;         // 4M
  const int NW = BINS * BINS;              // 1M
  f32_to_bf16_kernel<<<NX / 256, 256, 0, stream>>>(x,   xb,  NX);
  f32_to_bf16_kernel<<<NW / 256, 256, 0, stream>>>(q_w, wqb, NW);
  f32_to_bf16_kernel<<<NW / 256, 256, 0, stream>>>(k_w, wkb, NW);
  f32_to_bf16_kernel<<<NW / 256, 256, 0, stream>>>(v_w, wvb, NW);
  f32_to_bf16_kernel<<<NW / 256, 256, 0, stream>>>(o_w, wob, NW);

  dim3 gG(BINS / 256, SEQ / 64, NBC);      // block tile 64(M) x 256(N)
  const long aS = (long)SEQ * BINS, yS = (long)SEQ * BINS;

  // Q projection -> conv+RoPE -> [f][w][d]
  gemm_xwT_kernel<<<gG, 256, 0, stream>>>(xb, wqb, ysc, aS, yS, BINS, BINS);
  conv_rope_kernel<<<NX / 256, 256, 0, stream>>>(ysc, q_c, invf, qbf, 1, 0);
  // K projection -> conv+RoPE -> [f][w][d]
  gemm_xwT_kernel<<<gG, 256, 0, stream>>>(xb, wkb, ysc, aS, yS, BINS, BINS);
  conv_rope_kernel<<<NX / 256, 256, 0, stream>>>(ysc, k_c, invf, kbf, 1, 0);
  // V projection -> conv -> transposed [f][d][w]
  gemm_xwT_kernel<<<gG, 256, 0, stream>>>(xb, wvb, ysc, aS, yS, BINS, BINS);
  conv_rope_kernel<<<NX / 256, 256, 0, stream>>>(ysc, v_c, invf, vtb, 0, 1);

  // flash attention per (bc, head, 16-query tile)
  attn_fa_kernel<<<dim3(SEQ / 16, NBC * NHEAD), 32, 0, stream>>>(qbf, kbf, vtb, obf);

  // output projection, fp32 result straight to d_out
  gemm_xwT_kernel<<<gG, 256, 0, stream>>>(obf, wob, (float*)d_out, aS, yS, BINS, BINS);
}
